// STDP_30897994727567
// MI455X (gfx1250) — matrix operand order repaired and checked by reading, not compiled, not run on hardware
//
#include <hip/hip_runtime.h>

typedef __attribute__((ext_vector_type(16))) _Float16 v16h;
typedef __attribute__((ext_vector_type(8)))  _Float16 v8h;
typedef __attribute__((ext_vector_type(2)))  _Float16 h2;
typedef __attribute__((ext_vector_type(8)))  float    v8f;

#define T_STEPS 100
#define BATCH   32
#define N_IN    1024
#define N_OUT   512

#define MT 64     // out-rows per block
#define NT 128    // in-cols per block
#define PADH 40   // f16 operand column stride (80B: bank-stride 20, 16B aligned)

// A-fragment (16x32 MxK, 16-bit): arr is [col][b] f16, col = M, b = K.
// lane half h = lane/16: elements 0..7 -> K = h*8 + j ; 8..15 -> K = 16 + h*8 + (j-8)
__device__ __forceinline__ v16h loadA(const _Float16* arr, int col, int lane) {
  const _Float16* p = arr + col * PADH + ((lane >> 4) << 3);
  v8h x0 = *(const v8h*)p;          // K = h*8 + 0..7
  v8h x1 = *(const v8h*)(p + 16);   // K = 16 + h*8 + 0..7
  return __builtin_shufflevector(x0, x1, 0,1,2,3,4,5,6,7,8,9,10,11,12,13,14,15);
}

// B-fragment (32x16 KxN, 16-bit): arr is [col][b] f16, col = N, b = K.
// lane half h: elements 0..7 -> K = h*8 + j ; 8..15 -> K = 16 + h*8 + (j-8), N = lane%16
__device__ __forceinline__ v16h loadB(const _Float16* arr, int col, int lane) {
  const _Float16* p = arr + col * PADH + ((lane >> 4) << 3);
  v8h x0 = *(const v8h*)p;          // K = h*8 + 0..7
  v8h x1 = *(const v8h*)(p + 16);   // K = 16 + h*8 + 0..7
  return __builtin_shufflevector(x0, x1, 0,1,2,3,4,5,6,7,8,9,10,11,12,13,14,15);
}

__global__ __launch_bounds__(256) void stdp_wmma_kernel(
    const float* __restrict__ weight,       // [N_OUT, N_IN]
    const float* __restrict__ pre_spikes,   // [T, B, N_IN]
    const float* __restrict__ post_spikes,  // [T, B, N_OUT]
    const float* __restrict__ pre_trace0,   // [B, N_IN]
    const float* __restrict__ post_trace0,  // [B, N_OUT]
    float* __restrict__ out)                // dw | pre_tr | post_tr
{
  // double-buffered f16 operand arrays, [col][b] layout, b contiguous
  __shared__ alignas(16) _Float16 preS16[2][NT * PADH];
  __shared__ alignas(16) _Float16 preT16[2][NT * PADH];
  __shared__ alignas(16) _Float16 postS16[2][MT * PADH];
  __shared__ alignas(16) _Float16 postT16[2][MT * PADH];

  const int tid  = threadIdx.x;
  const int lane = tid & 31;
  const int wave = tid >> 5;
  const int oBase = blockIdx.x * MT;
  const int iBase = blockIdx.y * NT;

  const float decay = 0.95122942450071400910f;  // exp(-1/20)

  // Per-thread static (col, batch-pair) ownership:
  //  pre : 8 pairs -> i = (it*256+tid)&127, b = 2*((it*256+tid)>>7)
  //  post: 4 pairs -> o = (it*256+tid)&63,  b = 2*((it*256+tid)>>6)
  float preTrR[16];   // trace state in registers
  float postTrR[8];

  // ---- init trace registers from inputs ----
  #pragma unroll
  for (int it = 0; it < 8; ++it) {
    int idx = it * 256 + tid;
    int i = idx & (NT - 1);
    int b = (idx >> 7) << 1;
    preTrR[2 * it]     = pre_trace0[b * N_IN + iBase + i];
    preTrR[2 * it + 1] = pre_trace0[(b + 1) * N_IN + iBase + i];
  }
  #pragma unroll
  for (int it = 0; it < 4; ++it) {
    int idx = it * 256 + tid;
    int o = idx & (MT - 1);
    int b = (idx >> 6) << 1;
    postTrR[2 * it]     = post_trace0[b * N_OUT + oBase + o];
    postTrR[2 * it + 1] = post_trace0[(b + 1) * N_OUT + oBase + o];
  }

  // ---- staging helpers ----
  auto loadSpikes = [&](int t, float (&sPre)[16], float (&sPost)[8]) {
    const float* preRow  = pre_spikes  + (size_t)t * BATCH * N_IN  + iBase;
    const float* postRow = post_spikes + (size_t)t * BATCH * N_OUT + oBase;
    if (t + 1 < T_STEPS) {  // prefetch one step ahead of these loads
      __builtin_prefetch(preRow  + BATCH * N_IN  + (tid >> 3) * N_IN  + (tid & 7) * 16, 0, 1);
      __builtin_prefetch(postRow + BATCH * N_OUT + (tid >> 3) * N_OUT + (tid & 7) * 8,  0, 1);
    }
    #pragma unroll
    for (int it = 0; it < 8; ++it) {
      int idx = it * 256 + tid;
      int i = idx & (NT - 1);
      int b = (idx >> 7) << 1;
      sPre[2 * it]     = preRow[b * N_IN + i];
      sPre[2 * it + 1] = preRow[(b + 1) * N_IN + i];
    }
    #pragma unroll
    for (int it = 0; it < 4; ++it) {
      int idx = it * 256 + tid;
      int o = idx & (MT - 1);
      int b = (idx >> 6) << 1;
      sPost[2 * it]     = postRow[b * N_OUT + o];
      sPost[2 * it + 1] = postRow[(b + 1) * N_OUT + o];
    }
  };

  auto updateStore = [&](const float (&sPre)[16], const float (&sPost)[8],
                         _Float16* wPreS, _Float16* wPreT,
                         _Float16* wPostS, _Float16* wPostT) {
    #pragma unroll
    for (int it = 0; it < 8; ++it) {
      int idx = it * 256 + tid;
      int i = idx & (NT - 1);
      int b = (idx >> 7) << 1;
      float t0 = preTrR[2 * it]     * decay + sPre[2 * it];
      float t1 = preTrR[2 * it + 1] * decay + sPre[2 * it + 1];
      preTrR[2 * it] = t0;  preTrR[2 * it + 1] = t1;
      h2 ps; ps[0] = (_Float16)sPre[2 * it]; ps[1] = (_Float16)sPre[2 * it + 1];
      h2 pt; pt[0] = (_Float16)t0;           pt[1] = (_Float16)t1;
      *(h2*)(wPreS + i * PADH + b) = ps;     // packed ds_store_b32
      *(h2*)(wPreT + i * PADH + b) = pt;
    }
    #pragma unroll
    for (int it = 0; it < 4; ++it) {
      int idx = it * 256 + tid;
      int o = idx & (MT - 1);
      int b = (idx >> 6) << 1;
      float t0 = postTrR[2 * it]     * decay + sPost[2 * it];
      float t1 = postTrR[2 * it + 1] * decay + sPost[2 * it + 1];
      postTrR[2 * it] = t0;  postTrR[2 * it + 1] = t1;
      h2 ps; ps[0] = (_Float16)sPost[2 * it]; ps[1] = (_Float16)sPost[2 * it + 1];
      h2 pt; pt[0] = (_Float16)t0;            pt[1] = (_Float16)t1;
      *(h2*)(wPostS + o * PADH + b) = ps;
      *(h2*)(wPostT + o * PADH + b) = pt;
    }
  };

  // wave tile mapping: 4 (M) x 8 (N) grid of 16x16 tiles, 4 per wave
  const int m0w   = (wave & 3) << 4;
  const int nGrp  = (wave >> 2) << 6;
  const int mFrag = m0w + (lane & 15);

  v8f accP[4], accD[4];
  #pragma unroll
  for (int k = 0; k < 4; ++k) { accP[k] = {}; accD[k] = {}; }

  // one combined step: compute from read-buffer, stage t+1 into write-buffer
  auto step = [&](int t,
                  const _Float16* rPreS, const _Float16* rPreT,
                  const _Float16* rPostS, const _Float16* rPostT,
                  _Float16* wPreS, _Float16* wPreT,
                  _Float16* wPostS, _Float16* wPostT) {
    const int tn = t + 1;
    float sPre[16], sPost[8];
    if (tn < T_STEPS) loadSpikes(tn, sPre, sPost);   // long-latency first

    // WMMA phase (overlaps with the global loads above)
    v16h aS = loadA(rPostS, mFrag, lane);   // post spikes  (LTP A)
    v16h aT = loadA(rPostT, mFrag, lane);   // post traces  (LTD A)
    #pragma unroll
    for (int k = 0; k < 4; ++k) {
      int nFrag = nGrp + (k << 4) + (lane & 15);
      v16h bT = loadB(rPreT, nFrag, lane);  // pre traces (LTP B)
      v16h bS = loadB(rPreS, nFrag, lane);  // pre spikes (LTD B)
      accP[k] = __builtin_amdgcn_wmma_f32_16x16x32_f16(
          false, aS, false, bT, (short)0, accP[k], false, false);
      accD[k] = __builtin_amdgcn_wmma_f32_16x16x32_f16(
          false, aT, false, bS, (short)0, accD[k], false, false);
    }

    if (tn < T_STEPS) updateStore(sPre, sPost, wPreS, wPreT, wPostS, wPostT);
    __syncthreads();
  };

  // ---- prologue: stage t=0 into buffer 0 ----
  {
    float sPre[16], sPost[8];
    loadSpikes(0, sPre, sPost);
    updateStore(sPre, sPost, preS16[0], preT16[0], postS16[0], postT16[0]);
    __syncthreads();
  }

  // ---- main loop, unrolled x2 for static buffer selection ----
  for (int t = 0; t < T_STEPS; t += 2) {
    step(t,     preS16[0], preT16[0], postS16[0], postT16[0],
                preS16[1], preT16[1], postS16[1], postT16[1]);
    step(t + 1, preS16[1], preT16[1], postS16[1], postT16[1],
                preS16[0], preT16[0], postS16[0], postT16[0]);
  }

  // ---- epilogue: soft-bounded weighting, write dw ----
  const float sP =  1e-4f / 32.0f;
  const float sD = -1e-4f / 32.0f;
  const int rowBase = oBase + m0w + ((lane >> 4) << 3);  // C layout: lanes>=16 -> M+8
  #pragma unroll
  for (int k = 0; k < 4; ++k) {
    int n = iBase + nGrp + (k << 4) + (lane & 15);
    #pragma unroll
    for (int r = 0; r < 8; ++r) {
      size_t idx = (size_t)(rowBase + r) * N_IN + n;
      float w = weight[idx];
      out[idx] = sP * (1.0f - w) * accP[k][r] + sD * w * accD[k][r];
    }
  }

  // ---- final trace states written straight from registers (edge blocks) ----
  const size_t PRE_OFF  = (size_t)N_OUT * N_IN;              // 524288
  const size_t POST_OFF = PRE_OFF + (size_t)BATCH * N_IN;    // 557056
  if (blockIdx.x == 0) {
    #pragma unroll
    for (int it = 0; it < 8; ++it) {
      int idx = it * 256 + tid;
      int i = idx & (NT - 1);
      int b = (idx >> 7) << 1;
      out[PRE_OFF + (size_t)b * N_IN + iBase + i]       = preTrR[2 * it];
      out[PRE_OFF + (size_t)(b + 1) * N_IN + iBase + i] = preTrR[2 * it + 1];
    }
  }
  if (blockIdx.y == 0) {
    #pragma unroll
    for (int it = 0; it < 4; ++it) {
      int idx = it * 256 + tid;
      int o = idx & (MT - 1);
      int b = (idx >> 6) << 1;
      out[POST_OFF + (size_t)b * N_OUT + oBase + o]       = postTrR[2 * it];
      out[POST_OFF + (size_t)(b + 1) * N_OUT + oBase + o] = postTrR[2 * it + 1];
    }
  }
}

extern "C" void kernel_launch(void* const* d_in, const int* in_sizes, int n_in,
                              void* d_out, int out_size, void* d_ws, size_t ws_size,
                              hipStream_t stream) {
  const float* weight      = (const float*)d_in[0];
  const float* pre_spikes  = (const float*)d_in[1];
  const float* post_spikes = (const float*)d_in[2];
  const float* pre_trace   = (const float*)d_in[3];
  const float* post_trace  = (const float*)d_in[4];
  float* out = (float*)d_out;

  dim3 grid(N_OUT / MT, N_IN / NT);  // 8 x 8 = 64 blocks
  stdp_wmma_kernel<<<grid, 256, 0, stream>>>(
      weight, pre_spikes, post_spikes, pre_trace, post_trace, out);
}
